// GRU_59554016526678
// MI455X (gfx1250) — compile-verified, matrix operand order checked by reading, not run on hardware
//
#include <hip/hip_runtime.h>
#include <math.h>

#define BB 512
#define SS 256
#define DD 256
#define NSTEPS 256

typedef __attribute__((ext_vector_type(16))) __bf16 v16bf;
typedef __attribute__((ext_vector_type(8)))  float  v8f;

union V16 { v16bf b; unsigned short u[16]; };
union BFU { __bf16 b; unsigned short u; };

// native conversion: clang emits v_cvt_pk_bf16_f32
static __device__ __forceinline__ unsigned short f2bf(float f) {
  BFU c; c.b = (__bf16)f; return c.u;
}
static __device__ __forceinline__ float bf2f(unsigned short u) {
  return __uint_as_float(((unsigned int)u) << 16);
}

// hardware tanh if the gfx1250 builtin exists, else OCML fallback
static __device__ __forceinline__ float fast_tanh(float x) {
#if __has_builtin(__builtin_amdgcn_tanhf)
  return __builtin_amdgcn_tanhf(x);
#elif __has_builtin(__builtin_amdgcn_tanh_f32)
  return __builtin_amdgcn_tanh_f32(x);
#else
  return tanhf(x);
#endif
}

// K-index mapping for 16-bit A/B fragments (ISA 7.12.2, 16x32 bf16 layout):
// lanes 0-15 hold K {0..7,16..23}+..., lanes 16-31 hold K {8..15,24..31}+...
static __device__ __forceinline__ int kmap(int i, int lane) {
  return (i & 7) + ((i >> 3) << 4) + ((lane >> 4) << 3);
}

// A fragment from LDS bf16 [16][256] row-major
static __device__ __forceinline__ v16bf frag_a_bf_lds(const unsigned short* base, int kbase, int lane) {
  int m = lane & 15;
  V16 f;
#pragma unroll
  for (int i = 0; i < 16; ++i) f.u[i] = base[m * 256 + kbase + kmap(i, lane)];
  return f.b;
}
// A fragment from LDS fp32 [16][256] (native cvt to bf16)
static __device__ __forceinline__ v16bf frag_a_f32_lds(const float* base, int kbase, int lane) {
  int m = lane & 15;
  V16 f;
#pragma unroll
  for (int i = 0; i < 16; ++i) f.b[i] = (__bf16)base[m * 256 + kbase + kmap(i, lane)];
  return f.b;
}
// B fragment: rows n of a bf16 [N][256] weight (B[k][n] = W[n][k])
static __device__ __forceinline__ v16bf frag_b_bf_g(const unsigned short* W, int nbase, int kbase, int lane) {
  int n = nbase + (lane & 15);
  V16 f;
#pragma unroll
  for (int i = 0; i < 16; ++i) f.u[i] = W[(size_t)n * 256 + kbase + kmap(i, lane)];
  return f.b;
}
// fragment from fp32 global row-major [rows][256]
static __device__ __forceinline__ v16bf frag_f32_g(const float* W, int rbase, int kbase, int lane) {
  int r = rbase + (lane & 15);
  V16 f;
#pragma unroll
  for (int i = 0; i < 16; ++i) f.b[i] = (__bf16)W[(size_t)r * 256 + kbase + kmap(i, lane)];
  return f.b;
}

// ---------------- kernel 1: weights fp32 -> bf16 ----------------
__global__ void conv_weights_kernel(const float* __restrict__ Wa,
                                    const float* __restrict__ W_ih,
                                    const float* __restrict__ W_hh,
                                    unsigned short* __restrict__ Wa_bf,
                                    unsigned short* __restrict__ Wih_bf,
                                    unsigned short* __restrict__ Whh_bf) {
  int i = blockIdx.x * blockDim.x + threadIdx.x;  // over 768*256
  if (i >= 768 * 256) return;
  int n = i >> 8, k = i & 255;
  Whh_bf[i] = f2bf(W_hh[i]);
  Wih_bf[i] = f2bf(W_ih[(size_t)n * 259 + k]);   // first 256 cols of [768][259]
  if (n < 256) Wa_bf[i] = f2bf(Wa[i]);
}

// ------------- kernel 2: Ua_keys = bf16(e_all @ Ua^T + bu) -------------
// One wave per 16x16 output tile of the [B*S, 256] x [256,256] GEMM.
__global__ __launch_bounds__(256) void keys_gemm_kernel(const float* __restrict__ e_all,
                                                        const float* __restrict__ Ua,
                                                        const float* __restrict__ bu,
                                                        unsigned short* __restrict__ keys_bf) {
  int gwave = (blockIdx.x * blockDim.x + threadIdx.x) >> 5;
  int lane = threadIdx.x & 31;
  int mt = gwave >> 4;         // 8192 m-tiles of 16 rows
  int nt = gwave & 15;         // 16 n-tiles
  int mbase = mt * 16, nbase = nt * 16;
  v8f acc;
#pragma unroll
  for (int i = 0; i < 8; ++i) acc[i] = 0.f;
#pragma unroll
  for (int kk = 0; kk < 8; ++kk) {
    v16bf a = frag_f32_g(e_all, mbase, kk * 32, lane);
    v16bf b = frag_f32_g(Ua, nbase, kk * 32, lane);
    acc = __builtin_amdgcn_wmma_f32_16x16x32_bf16(false, a, false, b, (short)0, acc, false, false);
  }
  int n = nbase + (lane & 15);
  int mofs = (lane >> 4) * 8;
  float bn = bu[n];
#pragma unroll
  for (int r = 0; r < 8; ++r) {
    int m = mbase + mofs + r;
    keys_bf[(size_t)m * 256 + n] = f2bf(acc[r] + bn);
  }
}

// ------------- kernel 3: persistent autoregressive decode -------------
// 32 workgroups x 512 threads (16 waves). Workgroup owns 16 batch rows;
// wave w owns row w for the elementwise phases; loops over all 256 steps.
__global__ __launch_bounds__(512, 1) void decode_kernel(
    const float* __restrict__ e_all, const float* __restrict__ e_last,
    const float* __restrict__ ba, const float* __restrict__ Va, const float* __restrict__ bv,
    const float* __restrict__ W_ih, const float* __restrict__ b_ih, const float* __restrict__ b_hh,
    const float* __restrict__ Wo, const float* __restrict__ bo,
    const unsigned short* __restrict__ keys_bf,
    const unsigned short* __restrict__ Wa_bf,
    const unsigned short* __restrict__ Wih_bf,
    const unsigned short* __restrict__ Whh_bf,
    float* __restrict__ d_outputs,   // [B, STEPS, 3]
    float* __restrict__ hT,          // [B, D]
    float* __restrict__ cross_attn)  // [B, STEPS*S]
{
  __shared__ unsigned short hbf[16][256];       // h in bf16 (WMMA A operand)    8KB
  __shared__ float qc[16][256];                 // q, then ctx                  16KB
  __shared__ unsigned short gate[4][16][256];   // r,z,i_n,h_n bf16 / scores    32KB
  __shared__ float vva[256];                    // Va cache                      1KB
  __shared__ float xin[16][4];                  // autoregressive input
  float* sbuf = (float*)(&gate[0][0][0]);       // scores/weights [16][256] fp32 alias

  const int tid = threadIdx.x;
  const int lane = tid & 31;
  const int wave = tid >> 5;
  const int b0 = blockIdx.x * 16;

  // h0 from e_last; wave w owns row w, lane owns d = lane*8+j (fp32 in regs)
  float hreg[8];
  {
    const float* src = e_last + (size_t)(b0 + wave) * DD;
#pragma unroll
    for (int j = 0; j < 8; ++j) {
      int d = lane * 8 + j;
      hreg[j] = src[d];
      hbf[wave][d] = f2bf(hreg[j]);
    }
  }
  if (tid < 256) vva[tid] = Va[tid];
  if (tid < 64) xin[tid >> 2][tid & 3] = 0.f;
  __syncthreads();

  const float bvs = bv[0];

  for (int step = 0; step < NSTEPS; ++step) {
    // ---- phase 1: q = h @ Wa^T + ba ; wave w -> n-tile w*16 (WMMA)
    {
      int nb = wave * 16;
      v8f acc;
#pragma unroll
      for (int i = 0; i < 8; ++i) acc[i] = 0.f;
#pragma unroll
      for (int kk = 0; kk < 8; ++kk) {
        v16bf a = frag_a_bf_lds(&hbf[0][0], kk * 32, lane);
        v16bf b = frag_b_bf_g(Wa_bf, nb, kk * 32, lane);
        acc = __builtin_amdgcn_wmma_f32_16x16x32_bf16(false, a, false, b, (short)0, acc, false, false);
      }
      int n = nb + (lane & 15);
      int mofs = (lane >> 4) * 8;
#pragma unroll
      for (int r = 0; r < 8; ++r) qc[mofs + r][n] = acc[r] + ba[n];
    }
    __syncthreads();

    // ---- phase 2: scores[m][s] = Va . tanh(q[m] + Ua_keys[b,s]) + bv
    {
      int m = wave;
      const unsigned short* krow = keys_bf + (size_t)(b0 + m) * SS * DD;
#pragma unroll 1
      for (int c = 0; c < 8; ++c) {
        int s = lane + 32 * c;
        const unsigned short* kd = krow + (size_t)s * DD;
        if (c < 7) __builtin_prefetch(krow + (size_t)(s + 32) * DD, 0, 0);
        float acc = 0.f;
#pragma unroll 4
        for (int d = 0; d < 256; d += 2) {
          unsigned int pair = *(const unsigned int*)(kd + d);
          acc += fast_tanh(qc[m][d]     + bf2f((unsigned short)(pair & 0xFFFFu))) * vva[d];
          acc += fast_tanh(qc[m][d + 1] + bf2f((unsigned short)(pair >> 16)))     * vva[d + 1];
        }
        sbuf[m * 256 + s] = acc + bvs;
      }
    }
    __syncthreads();

    // ---- phase 3: softmax per row (wave-level shuffles) + emit cross_attn
    {
      int m = wave;
      float v[8];
      float mx = -1e30f;
#pragma unroll
      for (int c = 0; c < 8; ++c) { v[c] = sbuf[m * 256 + lane + 32 * c]; mx = fmaxf(mx, v[c]); }
#pragma unroll
      for (int o = 16; o > 0; o >>= 1) mx = fmaxf(mx, __shfl_xor(mx, o, 32));
      float sum = 0.f;
#pragma unroll
      for (int c = 0; c < 8; ++c) { v[c] = __expf(v[c] - mx); sum += v[c]; }
#pragma unroll
      for (int o = 16; o > 0; o >>= 1) sum += __shfl_xor(sum, o, 32);
      float inv = 1.f / sum;
      float* ca = cross_attn + (size_t)(b0 + m) * (NSTEPS * SS) + (size_t)step * SS;
#pragma unroll
      for (int c = 0; c < 8; ++c) {
        float w = v[c] * inv;
        sbuf[m * 256 + lane + 32 * c] = w;
        ca[lane + 32 * c] = w;
      }
    }
    __syncthreads();

    // ---- phase 4: ctx[m] = sum_s w[m][s] * e_all[b,s]  (overlay into qc)
    {
      int m = wave;
      const float* er = e_all + (size_t)(b0 + m) * SS * DD;
      float acc[8];
#pragma unroll
      for (int j = 0; j < 8; ++j) acc[j] = 0.f;
      for (int s = 0; s < SS; ++s) {
        float w = sbuf[m * 256 + s];
        const float* ed = er + (size_t)s * DD;
#pragma unroll
        for (int j = 0; j < 8; ++j) acc[j] += w * ed[lane + 32 * j];
      }
#pragma unroll
      for (int j = 0; j < 8; ++j) qc[m][lane + 32 * j] = acc[j];
    }
    __syncthreads();

    // ---- phase 5: gi = [ctx;x]@W_ih^T, gh = h@W_hh^T (WMMA) -> gates
    {
      // hoist loop-invariant A fragments (ctx and h) out of the tile loop:
      // each is reused for all 3 N-tiles this wave computes.
      v16bf afc[8], afh[8];
#pragma unroll
      for (int kk = 0; kk < 8; ++kk) {
        afc[kk] = frag_a_f32_lds(&qc[0][0], kk * 32, lane);
        afh[kk] = frag_a_bf_lds(&hbf[0][0], kk * 32, lane);
      }
#pragma unroll 1
      for (int t = wave; t < 48; t += 16) {
        int nb = t * 16;
        v8f ai, ah;
#pragma unroll
        for (int i = 0; i < 8; ++i) { ai[i] = 0.f; ah[i] = 0.f; }
#pragma unroll
        for (int kk = 0; kk < 8; ++kk) {
          v16bf b1 = frag_b_bf_g(Wih_bf, nb, kk * 32, lane);
          ai = __builtin_amdgcn_wmma_f32_16x16x32_bf16(false, afc[kk], false, b1, (short)0, ai, false, false);
          v16bf b2 = frag_b_bf_g(Whh_bf, nb, kk * 32, lane);
          ah = __builtin_amdgcn_wmma_f32_16x16x32_bf16(false, afh[kk], false, b2, (short)0, ah, false, false);
        }
        int n = nb + (lane & 15);
        int g = n >> 8;           // 0=r, 1=z, 2=n-gate
        int nn = n & 255;
        int mofs = (lane >> 4) * 8;
        const float wih0 = W_ih[(size_t)n * 259 + 256];
        const float wih1 = W_ih[(size_t)n * 259 + 257];
        const float wih2 = W_ih[(size_t)n * 259 + 258];
        const float bihn = b_ih[n], bhhn = b_hh[n];
#pragma unroll
        for (int r = 0; r < 8; ++r) {
          int m = mofs + r;
          float gi = ai[r] + bihn + xin[m][0] * wih0 + xin[m][1] * wih1 + xin[m][2] * wih2;
          float gh = ah[r] + bhhn;
          if (g == 0) {
            gate[0][m][nn] = f2bf(1.f / (1.f + __expf(-(gi + gh))));
          } else if (g == 1) {
            gate[1][m][nn] = f2bf(1.f / (1.f + __expf(-(gi + gh))));
          } else {
            gate[2][m][nn] = f2bf(gi);
            gate[3][m][nn] = f2bf(gh);
          }
        }
      }
    }
    __syncthreads();

    // ---- phase 6: h update + output linear (wave-reduced) + x_in feedback
    {
      int m = wave;
      float p0 = 0.f, p1 = 0.f, p2 = 0.f;
#pragma unroll
      for (int j = 0; j < 8; ++j) {
        int d = lane * 8 + j;
        float r  = bf2f(gate[0][m][d]);
        float z  = bf2f(gate[1][m][d]);
        float in_ = bf2f(gate[2][m][d]);
        float hn = bf2f(gate[3][m][d]);
        float nn = fast_tanh(in_ + r * hn);
        float h = (1.f - z) * nn + z * hreg[j];
        hreg[j] = h;
        hbf[m][d] = f2bf(h);
        p0 += h * Wo[0 * 256 + d];
        p1 += h * Wo[1 * 256 + d];
        p2 += h * Wo[2 * 256 + d];
      }
#pragma unroll
      for (int o = 16; o > 0; o >>= 1) {
        p0 += __shfl_xor(p0, o, 32);
        p1 += __shfl_xor(p1, o, 32);
        p2 += __shfl_xor(p2, o, 32);
      }
      if (lane == 0) {
        float o0 = p0 + bo[0], o1 = p1 + bo[1], o2 = p2 + bo[2];
        size_t base = ((size_t)(b0 + m) * NSTEPS + step) * 3;
        d_outputs[base + 0] = o0;
        d_outputs[base + 1] = o1;
        d_outputs[base + 2] = o2;
        xin[m][0] = o0; xin[m][1] = o1; xin[m][2] = o2;
      }
    }
    __syncthreads();
  }

  // final hidden state
  {
    int m = wave;
#pragma unroll
    for (int j = 0; j < 8; ++j) {
      int d = lane * 8 + j;
      hT[(size_t)(b0 + m) * DD + d] = hreg[j];
    }
  }
}

extern "C" void kernel_launch(void* const* d_in, const int* in_sizes, int n_in,
                              void* d_out, int out_size, void* d_ws, size_t ws_size,
                              hipStream_t stream) {
  const float* e_all  = (const float*)d_in[0];
  const float* e_last = (const float*)d_in[1];
  const float* Wa     = (const float*)d_in[2];
  const float* ba     = (const float*)d_in[3];
  const float* Ua     = (const float*)d_in[4];
  const float* bu     = (const float*)d_in[5];
  const float* Va     = (const float*)d_in[6];
  const float* bv     = (const float*)d_in[7];
  const float* W_ih   = (const float*)d_in[8];
  const float* b_ih   = (const float*)d_in[9];
  const float* W_hh   = (const float*)d_in[10];
  const float* b_hh   = (const float*)d_in[11];
  const float* Wo     = (const float*)d_in[12];
  const float* bo     = (const float*)d_in[13];

  // workspace layout (bf16): Ua_keys [B*S*D] + Wa_bf + Wih_bf + Whh_bf
  unsigned short* keys_bf = (unsigned short*)d_ws;
  unsigned short* Wa_bf   = keys_bf + (size_t)BB * SS * DD;
  unsigned short* Wih_bf  = Wa_bf + 256 * 256;
  unsigned short* Whh_bf  = Wih_bf + 768 * 256;

  float* d_outputs = (float*)d_out;                       // [B, STEPS, 3]
  float* hT        = d_outputs + (size_t)BB * NSTEPS * 3; // [1, B, D]
  float* cross     = hT + (size_t)BB * DD;                // [B, 1, STEPS*S]

  hipLaunchKernelGGL(conv_weights_kernel, dim3((768 * 256 + 255) / 256), dim3(256), 0, stream,
                     Wa, W_ih, W_hh, Wa_bf, Wih_bf, Whh_bf);
  // 8192 m-tiles * 16 n-tiles = 131072 waves; 8 waves (256 thr) per block
  hipLaunchKernelGGL(keys_gemm_kernel, dim3(16384), dim3(256), 0, stream,
                     e_all, Ua, bu, keys_bf);
  hipLaunchKernelGGL(decode_kernel, dim3(BB / 16), dim3(512), 0, stream,
                     e_all, e_last, ba, Va, bv, W_ih, b_ih, b_hh, Wo, bo,
                     keys_bf, Wa_bf, Wih_bf, Whh_bf, d_outputs, hT, cross);
}
